// CLSTM_43224550867776
// MI455X (gfx1250) — compile-verified, matrix operand order checked by reading, not run on hardware
//
#include <hip/hip_runtime.h>

// ---------------------------------------------------------------------------
// Complex LSTM (T=256, B=64, L=2, H=512) for MI455X / gfx1250.
//
//  * Complex clinear folded into real GEMM:  [hr|hi] @ [[wr, wi],[-wi, wr]]
//  * Weights packed once per launch into bf16 WMMA-B fragment tiles
//    (~34MB, resident in 192MB L2; steps re-read from L2, not HBM).
//  * Activations kept in bf16 in global memory; per-kernel staging into LDS
//    is done by the Tensor Data Mover (tensor_load_to_lds) with the TDM pad
//    feature generating the bank-conflict-avoiding LDS row stride (2080B).
//  * While the DMA runs, compute waves prefetch their weight-fragment streams
//    (global_prefetch) to warm the L2->WGP path.
//  * GEMMs: v_wmma_f32_16x16x32_bf16, f32 accumulate; gate nonlinearities,
//    c-state and complex cell update in f32.
//  * Per timestep: cell(L0) -> cell(L1) -> output projection, stream-ordered.
//    h ping-pongs (bf16); canonical f32 h/c kept for final state outputs.
// ---------------------------------------------------------------------------

typedef __attribute__((ext_vector_type(4)))  float        v4f;
typedef __attribute__((ext_vector_type(8)))  float        v8f;
typedef __attribute__((ext_vector_type(4)))  __bf16       v4bf;
typedef __attribute__((ext_vector_type(8)))  __bf16       v8bf;
typedef __attribute__((ext_vector_type(16))) __bf16       v16bf;
typedef __attribute__((ext_vector_type(4)))  unsigned int v4u;
typedef __attribute__((ext_vector_type(4)))  int          v4i;
typedef __attribute__((ext_vector_type(8)))  int          v8i;

#define HH     512          // H
#define HC     1024         // 2H
#define BATCH  64
#define TSTEPS 256
#define XSE    1040         // LDS row stride in bf16 elements (2080 bytes)
#define XROWB  2080         // LDS row stride bytes (2048 data + 2x16B TDM pad)
#define ABUF   (BATCH * XROWB)   // 133120 bytes per staged activation tile
#define GS     132          // LDS row stride (f32) for gate staging

// ---------------------------------------------------------------------------
// Toolchain probes (visible in compiler stderr):
// ---------------------------------------------------------------------------
#if defined(__has_builtin)
#if __has_builtin(__builtin_amdgcn_tensor_load_to_lds)
#define HAVE_TDM 1
#endif
#endif

#if defined(HAVE_TDM)
#pragma message("CLSTM probe: __builtin_amdgcn_tensor_load_to_lds AVAILABLE")
#if __has_include(<hip/amd_detail/amd_gfx1250_TDM.h>)
#pragma message("CLSTM probe: gfx1250 TDM header present -> using 6-arg builtin form")
#else
#pragma message("CLSTM probe: no gfx1250 TDM header -> using 5-arg builtin form")
#endif
#else
#pragma message("CLSTM probe: TDM builtin ABSENT -> cooperative LDS staging fallback")
#endif

// ---------------------------------------------------------------------------
// TDM: load a 64-row x 2048B bf16 tile from global into LDS, padding 16B
// after every 1024B so the LDS row stride becomes 2080B (16B aligned rows,
// ds_load_b128-legal fragment addresses, ~2-way bank conflicts).
// D# built per CDNA5 ISA sec 8.3/8.4:
//   group0: count=1 | lds_addr | global_addr[56:0] | type=2
//   group1: data_size=4B, pad_enable, pad_interval=7 (1024B), pad_amount=3
//           (4 dwords); tensor_dim0=512dw, tensor_dim1=64, tile=512x64,
//           tensor_dim0_stride=512dw
// ---------------------------------------------------------------------------
__device__ static inline void tdm_stage_tile(void* lds_ptr, const void* gptr)
{
#if defined(HAVE_TDM)
    unsigned lds_off = (unsigned)(unsigned long long)lds_ptr; // LDS aperture low bits
    unsigned long long ga = (unsigned long long)gptr;

    v4u g0;
    g0[0] = 1u;                                        // count=1, user desc
    g0[1] = lds_off;                                   // lds_addr
    g0[2] = (unsigned)(ga & 0xffffffffu);              // global_addr[31:0]
    g0[3] = (unsigned)((ga >> 32) & 0x01ffffffu) | (2u << 30); // [56:32] | type=2

    v8i g1;
    g1[0] = (2 << 16)            // data_size = 2 -> 4 bytes
          | (1 << 20)            // pad_enable
          | (7 << 22)            // pad_interval code 7 -> every 256 dw (1024B)
          | (3 << 25);           // pad_amount  code 3 -> 4 dw (16B)
    g1[1] = (int)(512u << 16);   // tensor_dim0[15:0]=512 at bits[63:48]
    g1[2] = (int)(64u << 16);    // tensor_dim0[31:16]=0 | tensor_dim1[15:0]=64
    g1[3] = (int)(512u << 16);   // tensor_dim1[31:16]=0 | tile_dim0=512
    g1[4] = 64;                  // tile_dim1=64 | tile_dim2=0
    g1[5] = 512;                 // tensor_dim0_stride[31:0] = 512 dw
    g1[6] = 0;                   // stride[47:32] | tensor_dim1_stride[15:0]
    g1[7] = 0;

    v4i gz = {0, 0, 0, 0};
#if __has_include(<hip/amd_detail/amd_gfx1250_TDM.h>)
    v8i gz8 = {0, 0, 0, 0, 0, 0, 0, 0};
    __builtin_amdgcn_tensor_load_to_lds(g0, g1, gz, gz, gz8, 0);
#else
    __builtin_amdgcn_tensor_load_to_lds(g0, g1, gz, gz, 0);
#endif
    __builtin_amdgcn_s_wait_tensorcnt(0);
#endif
}

// Fallback / generic cooperative staging (used when TDM builtin is absent):
// copy bf16 rows into the padded LDS layout.
__device__ static inline void coop_stage_tile(__bf16* lds, const __bf16* g, int tid)
{
    for (int i = tid; i < BATCH * 128; i += 256) {      // 8 bf16 per item
        int row = i >> 7;
        int col = (i & 127) * 8;
        int dst = row * XSE + col + ((col >= 512) ? 8 : 0);
        *(v8bf*)(lds + dst) = *(const v8bf*)(g + row * HC + col);
    }
}

// ---------------------------------------------------------------------------
// Pack (2,H,H) complex weight pairs into bf16 WMMA-B fragment order.
// frag(m, tn, tk): 512 bf16 at ((m*64 + tn)*32 + tk)*512; lane l owns 16
// contiguous bf16 at l*16 covering column n = tn*16 + (l&15) with K pattern
// lanes 0-15: K={0..7,16..23}, lanes 16-31: K={8..15,24..31}.
// Real matrix: Wcat[k][n]:  n<H -> ( k<H ?  wr[k][n]    : -wi[k-H][n]    )
//                           n>=H -> ( k<H ?  wi[k][n-H] :  wr[k-H][n-H] )
// ---------------------------------------------------------------------------
__global__ __launch_bounds__(256) void clstm_pack_kernel(
    const float* __restrict__ src, __bf16* __restrict__ dst, int nmat)
{
    long long d = (long long)blockIdx.x * 256 + threadIdx.x;
    long long total = (long long)nmat * HC * HC;
    if (d >= total) return;

    int r        = (int)(d & 511);
    long long fr = d >> 9;
    int lane = r >> 4, e = r & 15;
    int tk = (int)(fr & 31);
    int tn = (int)((fr >> 5) & 63);
    int m  = (int)(fr >> 11);

    int hi = lane >> 4, nl = lane & 15;
    int vv = e >> 1, p = e & 1;
    int kl = (vv < 4) ? ((hi ? 8 : 0) + 2 * vv + p)
                      : ((hi ? 24 : 16) + 2 * (vv - 4) + p);
    int k = tk * 32 + kl;
    int n = tn * 16 + nl;

    const float* wr = src + (long long)m * 2 * HH * HH;
    const float* wi = wr + HH * HH;
    float val;
    if (n < HH) {
        val = (k < HH) ? wr[k * HH + n] : -wi[(k - HH) * HH + n];
    } else {
        int oc = n - HH;
        val = (k < HH) ? wi[k * HH + oc] : wr[(k - HH) * HH + oc];
    }
    dst[d] = (__bf16)val;
}

// f32 -> bf16 bulk conversion (x input, h0 init)
__global__ __launch_bounds__(256) void clstm_cvt_kernel(
    const float* __restrict__ src, __bf16* __restrict__ dst, long long n4)
{
    long long i = (long long)blockIdx.x * 256 + threadIdx.x;
    if (i >= n4) return;
    v4f s = ((const v4f*)src)[i];
    v4bf b;
    b[0] = (__bf16)s[0]; b[1] = (__bf16)s[1];
    b[2] = (__bf16)s[2]; b[3] = (__bf16)s[3];
    ((v4bf*)dst)[i] = b;
}

// ---------------------------------------------------------------------------
// One LSTM cell step for one layer.
// Grid: 32 workgroups (wg owns 16 complex columns across all 4 gates),
// 256 threads = 8 waves.  Wave w: gate g=w>>1, imag-half im=w&1.
// gates(64x128) = h_prev @ Wfrag + x_in @ Ufrag  (K=1024 each, bf16 WMMA)
// then bias + sigmoid/tanh, LDS staging, fused complex cell update.
// ---------------------------------------------------------------------------
__global__ __launch_bounds__(256) void clstm_cell_kernel(
    const __bf16* __restrict__ xbf,      // 64 x 1024 bf16 (input acts)
    const __bf16* __restrict__ hprev,    // 64 x 1024 bf16
    __bf16*       __restrict__ hbf_out,  // 64 x 1024 bf16 (next-step GEMM A)
    float*        __restrict__ hf32_out, // 64 x 1024 f32  (final-state copy)
    float*        __restrict__ c_state,  // 64 x 1024 f32 (in-place)
    const __bf16* __restrict__ Ufrag,
    const __bf16* __restrict__ Wfrag,
    const float*  __restrict__ ub,
    const float*  __restrict__ wb)
{
    extern __shared__ char smem[];
    __bf16* xs = (__bf16*)smem;                 // 64 x XSE bf16 (padded)
    __bf16* hs = (__bf16*)(smem + ABUF);        // 64 x XSE bf16 (padded)
    float*  gs = (float*)(smem + 2 * ABUF);     // 64 x GS f32 (gates)

    const int tid  = threadIdx.x;
    const int wave = tid >> 5;
    const int lane = tid & 31;
    const int wg   = blockIdx.x;                // 0..31

    const int g  = wave >> 1;        // gate: 0=f 1=i 2=a 3=o
    const int im = wave & 1;         // 0=real half, 1=imag half
    const int tn = im * 32 + wg;
    const int nl = lane & 15;
    const int hi = lane >> 4;

    const __bf16* ufr = Ufrag + ((long long)(g * 64 + tn) * 32) * 512 + lane * 16;
    const __bf16* wfr = Wfrag + ((long long)(g * 64 + tn) * 32) * 512 + lane * 16;

#if defined(HAVE_TDM)
    if (wave == 0)      tdm_stage_tile(xs, xbf);
    else if (wave == 1) tdm_stage_tile(hs, hprev);
    else {
        // warm the L2->WGP path for this wave's weight streams while DMA runs
#pragma unroll
        for (int pf = 0; pf < 8; ++pf) {
            __builtin_prefetch(ufr + pf * 2048, 0, 0);   // every 4KB of stream
            __builtin_prefetch(wfr + pf * 2048, 0, 0);
        }
    }
#else
#pragma unroll
    for (int pf = 0; pf < 4; ++pf) {
        __builtin_prefetch(ufr + pf * 4096, 0, 0);
        __builtin_prefetch(wfr + pf * 4096, 0, 0);
    }
    coop_stage_tile(xs, xbf, tid);
    coop_stage_tile(hs, hprev, tid);
#endif
    __syncthreads();

    v8f acc[4] = {};

    // ---- GEMM 1: x @ U ----
    for (int tk = 0; tk < 32; ++tk) {
        union { v16bf v; v8bf h2[2]; } bf;
        bf.h2[0] = *(const v8bf*)(ufr + tk * 512);
        bf.h2[1] = *(const v8bf*)(ufr + tk * 512 + 8);
        const int cofs = tk * 32 + hi * 8 + ((tk >= 16) ? 8 : 0); // TDM pad skip
#pragma unroll
        for (int rt = 0; rt < 4; ++rt) {
            union { v16bf v; v8bf h2[2]; } af;
            const __bf16* arow = xs + (rt * 16 + nl) * XSE + cofs;
            af.h2[0] = *(const v8bf*)(arow);
            af.h2[1] = *(const v8bf*)(arow + 16);
            acc[rt] = __builtin_amdgcn_wmma_f32_16x16x32_bf16(
                false, af.v, false, bf.v, (short)0, acc[rt], false, false);
        }
    }
    // ---- GEMM 2: h_prev @ W ----
    for (int tk = 0; tk < 32; ++tk) {
        union { v16bf v; v8bf h2[2]; } bf;
        bf.h2[0] = *(const v8bf*)(wfr + tk * 512);
        bf.h2[1] = *(const v8bf*)(wfr + tk * 512 + 8);
        const int cofs = tk * 32 + hi * 8 + ((tk >= 16) ? 8 : 0);
#pragma unroll
        for (int rt = 0; rt < 4; ++rt) {
            union { v16bf v; v8bf h2[2]; } af;
            const __bf16* arow = hs + (rt * 16 + nl) * XSE + cofs;
            af.h2[0] = *(const v8bf*)(arow);
            af.h2[1] = *(const v8bf*)(arow + 16);
            acc[rt] = __builtin_amdgcn_wmma_f32_16x16x32_bf16(
                false, af.v, false, bf.v, (short)0, acc[rt], false, false);
        }
    }

    // ---- bias + nonlinearity, stage gates to LDS ----
    const int colg = im * HH + wg * 16 + nl;
    const float bias = ub[g * HC + colg] + wb[g * HC + colg];
#pragma unroll
    for (int rt = 0; rt < 4; ++rt) {
#pragma unroll
        for (int vv = 0; vv < 8; ++vv) {
            float pre = acc[rt][vv] + bias;
            float act = (g == 2) ? tanhf(pre)
                                 : 1.0f / (1.0f + __expf(-pre));
            gs[(rt * 16 + vv + 8 * hi) * GS + wave * 16 + nl] = act;
        }
    }
    __syncthreads();

    // ---- fused complex cell update: 64 rows x 16 complex columns ----
    for (int i = tid; i < BATCH * 16; i += 256) {
        int row = i >> 4, u = i & 15;
        const float* gr = gs + row * GS;
        float fr = gr[u],       fi = gr[16 + u];
        float ir = gr[32 + u],  ii = gr[48 + u];
        float ar = gr[64 + u],  ai = gr[80 + u];
        float pr = gr[96 + u],  pi = gr[112 + u];   // o gate
        int jc = wg * 16 + u;
        float cr = c_state[row * HC + jc];
        float ci = c_state[row * HC + HH + jc];
        float ctr = cr * fr - ci * fi + ar * ir - ai * ii;
        float cti = cr * fi + ci * fr + ar * ii + ai * ir;
        float tr = tanhf(ctr), ti = tanhf(cti);
        c_state[row * HC + jc]      = ctr;
        c_state[row * HC + HH + jc] = cti;
        float hr = pr * tr - pi * ti;
        float hi2 = pr * ti + pi * tr;
        hf32_out[row * HC + jc]      = hr;
        hf32_out[row * HC + HH + jc] = hi2;
        hbf_out[row * HC + jc]       = (__bf16)hr;
        hbf_out[row * HC + HH + jc]  = (__bf16)hi2;
    }
}

// ---------------------------------------------------------------------------
// Output projection: y = tanh(h @ WoutCat + bout).
// Grid: 8 workgroups x 8 waves -> 64 n-tiles, K = 1024.
// ---------------------------------------------------------------------------
__global__ __launch_bounds__(256) void clstm_out_kernel(
    const __bf16* __restrict__ hbf,
    const __bf16* __restrict__ Ofrag,
    const float*  __restrict__ bout,
    float*        __restrict__ y)
{
    extern __shared__ char smem[];
    __bf16* hsb = (__bf16*)smem;                // 64 x XSE bf16 (padded)

    const int tid  = threadIdx.x;
    const int wave = tid >> 5;
    const int lane = tid & 31;
    const int tn   = blockIdx.x * 8 + wave;

    const __bf16* ofr = Ofrag + ((long long)tn * 32) * 512 + lane * 16;

#if defined(HAVE_TDM)
    if (wave == 0) tdm_stage_tile(hsb, hbf);
    else {
#pragma unroll
        for (int pf = 0; pf < 8; ++pf)
            __builtin_prefetch(ofr + pf * 2048, 0, 0);
    }
#else
#pragma unroll
    for (int pf = 0; pf < 4; ++pf)
        __builtin_prefetch(ofr + pf * 4096, 0, 0);
    coop_stage_tile(hsb, hbf, tid);
#endif
    __syncthreads();

    const int nl = lane & 15;
    const int hi = lane >> 4;
    v8f acc[4] = {};

    for (int tk = 0; tk < 32; ++tk) {
        union { v16bf v; v8bf h2[2]; } bf;
        bf.h2[0] = *(const v8bf*)(ofr + tk * 512);
        bf.h2[1] = *(const v8bf*)(ofr + tk * 512 + 8);
        const int cofs = tk * 32 + hi * 8 + ((tk >= 16) ? 8 : 0);
#pragma unroll
        for (int rt = 0; rt < 4; ++rt) {
            union { v16bf v; v8bf h2[2]; } af;
            const __bf16* arow = hsb + (rt * 16 + nl) * XSE + cofs;
            af.h2[0] = *(const v8bf*)(arow);
            af.h2[1] = *(const v8bf*)(arow + 16);
            acc[rt] = __builtin_amdgcn_wmma_f32_16x16x32_bf16(
                false, af.v, false, bf.v, (short)0, acc[rt], false, false);
        }
    }

    const int col = tn * 16 + nl;
    const float b = bout[col];
#pragma unroll
    for (int rt = 0; rt < 4; ++rt) {
#pragma unroll
        for (int vv = 0; vv < 8; ++vv) {
            int row = rt * 16 + vv + 8 * hi;
            y[row * HC + col] = tanhf(acc[rt][vv] + b);
        }
    }
}

// ---------------------------------------------------------------------------
extern "C" void kernel_launch(void* const* d_in, const int* in_sizes, int n_in,
                              void* d_out, int out_size, void* d_ws, size_t ws_size,
                              hipStream_t stream)
{
    const float* x    = (const float*)d_in[0];   // (T, B, 2H)
    const float* h0   = (const float*)d_in[1];   // (L, B, 2H)
    const float* c0   = (const float*)d_in[2];   // (L, B, 2H)
    const float* Uw   = (const float*)d_in[3];   // (L, 4, 2, H, H)
    const float* Ubb  = (const float*)d_in[4];   // (L, 4, 2H)
    const float* Ww   = (const float*)d_in[5];   // (L, 4, 2, H, H)
    const float* Wbb  = (const float*)d_in[6];   // (L, 4, 2H)
    const float* Wout = (const float*)d_in[7];   // (2, H, H)
    const float* bout = (const float*)d_in[8];   // (2H,)
    float* out = (float*)d_out;                  // y | hn | cn

    const long long MAT   = (long long)HC * HC;  // elems per packed matrix
    const int       STATE = BATCH * HC;          // 65536 elems per layer state

    // workspace layout (~71 MB)
    char* ws = (char*)d_ws;
    __bf16* Ufrag = (__bf16*)ws;                       ws += 8 * MAT * 2;
    __bf16* Wfrag = (__bf16*)ws;                       ws += 8 * MAT * 2;
    __bf16* Ofrag = (__bf16*)ws;                       ws += MAT * 2;
    __bf16* xbf   = (__bf16*)ws;                       ws += (long long)TSTEPS * STATE * 2;
    __bf16* hbf   = (__bf16*)ws;                       ws += 4LL * STATE * 2;   // [2][L]
    float*  hf32  = (float*)ws;                        ws += 2LL * STATE * 4;   // [L]
    float*  cbuf  = (float*)ws;                        ws += 2LL * STATE * 4;   // [L]

    // ---- pack weights into bf16 WMMA fragment order ----
    clstm_pack_kernel<<<(int)(8 * MAT / 256), 256, 0, stream>>>(Uw, Ufrag, 8);
    clstm_pack_kernel<<<(int)(8 * MAT / 256), 256, 0, stream>>>(Ww, Wfrag, 8);
    clstm_pack_kernel<<<(int)(MAT / 256), 256, 0, stream>>>(Wout, Ofrag, 1);

    // ---- convert activations to bf16; init states ----
    clstm_cvt_kernel<<<(int)((long long)TSTEPS * STATE / 4 / 256), 256, 0, stream>>>(
        x, xbf, (long long)TSTEPS * STATE / 4);
    clstm_cvt_kernel<<<(int)(2LL * STATE / 4 / 256), 256, 0, stream>>>(
        h0, hbf, 2LL * STATE / 4);                         // ping buffer 0
    hipMemcpyAsync(cbuf, c0, 2ull * STATE * sizeof(float),
                   hipMemcpyDeviceToDevice, stream);

    const size_t SM_CELL = 2 * (size_t)ABUF + (size_t)(BATCH * GS) * sizeof(float);
    const size_t SM_OUT  = (size_t)ABUF;
    const long long LSTRIDE = 4 * MAT;           // packed elems per layer

    for (int t = 0; t < TSTEPS; ++t) {
        int p = t & 1;
        __bf16* cur = hbf + p * 2 * STATE;
        __bf16* nxt = hbf + (1 - p) * 2 * STATE;

        // layer 0
        clstm_cell_kernel<<<32, 256, SM_CELL, stream>>>(
            xbf + (long long)t * STATE, cur, nxt, hf32, cbuf,
            Ufrag, Wfrag, Ubb, Wbb);

        // layer 1 (input = layer 0's fresh h)
        clstm_cell_kernel<<<32, 256, SM_CELL, stream>>>(
            nxt, cur + STATE, nxt + STATE, hf32 + STATE, cbuf + STATE,
            Ufrag + LSTRIDE, Wfrag + LSTRIDE, Ubb + 4 * HC, Wbb + 4 * HC);

        // output projection
        clstm_out_kernel<<<8, 256, SM_OUT, stream>>>(
            nxt + STATE, Ofrag, bout, out + (long long)t * STATE);
    }

    // final states (f32 canonical copies)
    hipMemcpyAsync(out + (long long)TSTEPS * STATE, hf32,
                   2ull * STATE * sizeof(float), hipMemcpyDeviceToDevice, stream);
    hipMemcpyAsync(out + (long long)TSTEPS * STATE + 2 * STATE, cbuf,
                   2ull * STATE * sizeof(float), hipMemcpyDeviceToDevice, stream);
}
